// OnsetsAndFrames_self_attention_48189533061164
// MI455X (gfx1250) — matrix operand order repaired and checked by reading, not compiled, not run on hardware
//
#include <hip/hip_runtime.h>
#include <hip/hip_bf16.h>

typedef __bf16 bf16_t;
typedef __attribute__((ext_vector_type(16))) __bf16 v16bf;
typedef __attribute__((ext_vector_type(8)))  __bf16 v8bf;
typedef __attribute__((ext_vector_type(8)))  float  v8f;

#define WMMA_BF16(a,b,c) __builtin_amdgcn_wmma_f32_16x16x32_bf16(false,(a),false,(b),(short)0,(c),false,false)

// ---- CDNA5 async global->LDS copy (ASYNCcnt path), inline asm for toolchain portability ----
__device__ __forceinline__ void oaf_async_b128(unsigned lds_addr, const bf16_t* gaddr){
  asm volatile("global_load_async_to_lds_b128 %0, %1, off" :: "v"(lds_addr), "v"(gaddr) : "memory");
}
__device__ __forceinline__ void oaf_wait_async(){
  asm volatile("s_wait_asynccnt 0x0" ::: "memory");
}

// ---------------- small prep kernels ----------------

__global__ __launch_bounds__(128) void oaf_bnprep(const float* g, const float* be,
    const float* m, const float* v, const float* cb, float* alpha, float* beta, int C){
  int c = threadIdx.x;
  if (c < C){ float a = g[c]*rsqrtf(v[c]+1e-5f); alpha[c]=a; beta[c]=(cb[c]-m[c])*a+be[c]; }
}

// stage f32 [N,Ksrc] -> bf16 [Npad,Kp] zero-filled
__global__ __launch_bounds__(256) void oaf_stagew(const float* __restrict__ src, bf16_t* __restrict__ dst,
    int N, int Ksrc, int Kp, unsigned long long total){
  unsigned long long idx = (unsigned long long)blockIdx.x*256 + threadIdx.x;
  if (idx >= total) return;
  int k = (int)(idx % (unsigned)Kp); unsigned long long n = idx / (unsigned)Kp;
  float v = (n < (unsigned long long)N && k < Ksrc) ? src[n*(unsigned long long)Ksrc + k] : 0.f;
  dst[idx] = (bf16_t)v;
}

// conv weights [Co,Ci,3,3] f32 -> bf16 [9][CoPad][64] (ci contiguous)
__global__ __launch_bounds__(256) void oaf_stagecw(const float* __restrict__ src, bf16_t* __restrict__ dst,
    int Co, int Ci, int CoPad){
  unsigned long long idx = (unsigned long long)blockIdx.x*256 + threadIdx.x; // 9*CoPad*64 exact
  int ci = (int)(idx & 63); unsigned long long r = idx >> 6;
  int co = (int)(r % (unsigned)CoPad); int tap = (int)(r / (unsigned)CoPad);
  float v = (co < Co && ci < Ci) ? src[(unsigned long long)(co*Ci + ci)*9 + tap] : 0.f;
  dst[idx] = (bf16_t)v;
}

// conv1 (Cin=1) + BN + ReLU -> channels-last bf16 [B,512,229,64], pad channels zero
__global__ __launch_bounds__(256) void oaf_conv1(const float* __restrict__ spec, const float* __restrict__ w1,
    const float* __restrict__ alpha, const float* __restrict__ beta, bf16_t* __restrict__ Y){
  unsigned long long idx = (unsigned long long)blockIdx.x*256 + threadIdx.x;
  int c = (int)(idx & 63); unsigned long long m = idx >> 6;
  int x = (int)(m % 229u); unsigned long long bt = m / 229u;
  int t = (int)(bt % 512u); int b = (int)(bt / 512u);
  float val = 0.f;
  if (c < 48){
    float s = 0.f;
    #pragma unroll
    for (int dy=0; dy<3; dy++){
      int tt = t + dy - 1; if (tt < 0 || tt >= 512) continue;
      #pragma unroll
      for (int dx=0; dx<3; dx++){
        int xx = x + dx - 1; if (xx < 0 || xx >= 229) continue;
        s += spec[((unsigned long long)(b*512+tt))*229 + xx] * w1[c*9 + dy*3 + dx];
      }
    }
    val = fmaxf(0.f, s*alpha[c] + beta[c]);
  }
  Y[idx] = (bf16_t)val;
}

// ---------------- WMMA implicit-GEMM 3x3 conv + BN + ReLU ----------------
// X: bf16 [B,H,Wd,64] (Ci padded to 64), Wst: bf16 [9][CoPad][64]
// Y: bf16 [B,H,Wd,CoPad]; grid = (B*H*Wd/64, CoPad/32), block 256 (8 waves, 4Mx2N of 16x16)
__global__ __launch_bounds__(256) void oaf_conv3x3(const bf16_t* __restrict__ X, const bf16_t* __restrict__ Wst,
    const float* __restrict__ alpha, const float* __restrict__ beta, bf16_t* __restrict__ Y,
    int H, int Wd, int Co, int CoPad){
  int wave = threadIdx.x >> 5, lane = threadIdx.x & 31;
  int wm = wave & 3, wn = wave >> 2;
  int mBase = blockIdx.x*64 + wm*16;
  int nBase = blockIdx.y*32 + wn*16;
  int r = lane & 15, hv = lane >> 4;
  int m = mBase + r;
  int x = m % Wd; int bt = m / Wd; int t = bt % H; int b = bt / H;
  v8f acc = {};
  for (int tap = 0; tap < 9; tap++){
    int dy = tap/3 - 1, dx = tap - (tap/3)*3 - 1;
    int tt = t + dy, xx = x + dx;
    bool ok = (tt >= 0) && (tt < H) && (xx >= 0) && (xx < Wd);
    long long off = ok ? ((long long)((b*H + tt)*Wd + xx))*64 : 0;
    const bf16_t* px = X + off + (hv<<4);
    const bf16_t* pw = Wst + ((unsigned long long)(tap*CoPad + nBase + r))*64 + (hv<<4);
    #pragma unroll
    for (int ks = 0; ks < 2; ks++){
      v16bf af = {};
      if (ok){
        v8bf a0 = *(const v8bf*)(px + ks*32);
        v8bf a1 = *(const v8bf*)(px + ks*32 + 8);
        #pragma unroll
        for (int i=0;i<8;i++){ af[i]=a0[i]; af[8+i]=a1[i]; }
      }
      v16bf bfr = *(const v16bf*)(pw + ks*32);
      acc = WMMA_BF16(af, bfr, acc);
    }
  }
  int co = nBase + r;
  int row0 = mBase + (hv<<3);
  #pragma unroll
  for (int j=0;j<8;j++){
    int row = row0 + j;
    float v = 0.f;
    if (co < Co) v = fmaxf(0.f, acc[j]*alpha[co] + beta[co]);
    Y[(unsigned long long)row*CoPad + co] = (bf16_t)v;
  }
}

// max-pool over last spatial dim (stride 2), channels-last bf16
__global__ __launch_bounds__(256) void oaf_pool(const bf16_t* __restrict__ X, bf16_t* __restrict__ Y,
    int Wd, int Wo, int C){
  unsigned long long idx = (unsigned long long)blockIdx.x*256 + threadIdx.x;
  int c = (int)(idx % (unsigned)C); unsigned long long r1 = idx / (unsigned)C;
  int xo = (int)(r1 % (unsigned)Wo); unsigned long long bh = r1 / (unsigned)Wo;
  const bf16_t* p = X + (bh*Wd + 2*xo)*(unsigned long long)C + c;
  float a = (float)p[0], b = (float)p[C];
  Y[idx] = (bf16_t)fmaxf(a,b);
}

// pool2 + pack into FC-A layout: X [2048,114,96] -> A[m][c*57+xo]
__global__ __launch_bounds__(256) void oaf_pool2(const bf16_t* __restrict__ X, bf16_t* __restrict__ A){
  unsigned long long idx = (unsigned long long)blockIdx.x*256 + threadIdx.x; // 2048*96*57
  int xo = (int)(idx % 57u); unsigned long long r1 = idx / 57u;
  int c = (int)(r1 % 96u); unsigned long long m = r1 / 96u;
  const bf16_t* p = X + (m*114 + 2*xo)*96ULL + c;
  float a = (float)p[0], b = (float)p[96];
  A[m*5472ULL + (unsigned long long)c*57 + xo] = (bf16_t)fmaxf(a,b);
}

// ---------------- generic WMMA GEMM: C = act(A[M,K] * W[N,K]^T + bias) ----------------
// grid = (M/128, Npad/64); block 256 = 8 waves (4M x 2N), each wave a 32x32 tile (2x2 WMMA).
// B (weights) tile staged to LDS via async global->LDS loads, double-buffered.
__global__ __launch_bounds__(256) void oaf_gemm(const bf16_t* __restrict__ A, const bf16_t* __restrict__ W,
    const float* __restrict__ bias, float* __restrict__ outF, int ldoF,
    bf16_t* __restrict__ outBF, int ldoBF, int N, int K, int act){
  __shared__ __align__(64) bf16_t sB[2][64][32];
  int tid = threadIdx.x;
  int wave = tid >> 5, lane = tid & 31;
  int wm = wave & 3, wn = wave >> 2;
  int mBase = blockIdx.x*128 + wm*32;
  int nBase = blockIdx.y*64  + wn*32;
  int r = lane & 15, hv = lane >> 4;

  // async staging role: each thread copies one 16B chunk of the 64x32 B tile per k-step
  int n_l = tid >> 2, kc = tid & 3;
  const bf16_t* gB = W + (unsigned long long)(blockIdx.y*64 + n_l)*K + kc*8;
  unsigned lds0 = (unsigned)(unsigned long long)&sB[0][n_l][kc*8];
  unsigned lds1 = (unsigned)(unsigned long long)&sB[1][n_l][kc*8];

  const bf16_t* pa0 = A + (unsigned long long)(mBase + r)*K + (hv<<4);
  const bf16_t* pa1 = A + (unsigned long long)(mBase + 16 + r)*K + (hv<<4);

  v8f acc00 = {}, acc01 = {}, acc10 = {}, acc11 = {};

  oaf_async_b128(lds0, gB);       // prologue: k-step 0 into buffer 0
  oaf_wait_async();
  __syncthreads();

  int buf = 0;
  for (int kt = 0; kt < K; kt += 32){
    if (kt + 32 < K) oaf_async_b128(buf ? lds0 : lds1, gB + kt + 32);

    v8bf a00 = *(const v8bf*)(pa0 + kt);
    v8bf a01 = *(const v8bf*)(pa0 + kt + 8);
    v8bf a10 = *(const v8bf*)(pa1 + kt);
    v8bf a11 = *(const v8bf*)(pa1 + kt + 8);
    v16bf af0, af1;
    #pragma unroll
    for (int i=0;i<8;i++){ af0[i]=a00[i]; af0[8+i]=a01[i]; af1[i]=a10[i]; af1[8+i]=a11[i]; }
    v16bf bf0 = *(const v16bf*)&sB[buf][wn*32 + r][hv<<4];
    v16bf bf1 = *(const v16bf*)&sB[buf][wn*32 + 16 + r][hv<<4];

    acc00 = WMMA_BF16(af0, bf0, acc00);
    acc01 = WMMA_BF16(af0, bf1, acc01);
    acc10 = WMMA_BF16(af1, bf0, acc10);
    acc11 = WMMA_BF16(af1, bf1, acc11);

    oaf_wait_async();   // next buffer's LDS writes (this wave's) complete
    __syncthreads();    // all waves done reading current buffer + all waits done
    buf ^= 1;
  }

  int row0 = mBase + (hv<<3);
  #pragma unroll
  for (int ms = 0; ms < 2; ms++){
    #pragma unroll
    for (int ns = 0; ns < 2; ns++){
      const v8f& acc = ms ? (ns ? acc11 : acc10) : (ns ? acc01 : acc00);
      int col = nBase + ns*16 + r;
      if (col < N){
        float bv = bias ? bias[col] : 0.f;
        #pragma unroll
        for (int j=0;j<8;j++){
          int row = row0 + ms*16 + j;
          float v = acc[j] + bv;
          if (act == 1) v = 1.f/(1.f + __expf(-v));
          if (outF)  outF[(unsigned long long)row*ldoF + col] = v;
          if (outBF) outBF[(unsigned long long)row*ldoBF + col] = (bf16_t)v;
        }
      }
    }
  }
}

// pad rows: src [B*512,KD] -> dst [B*544,KD] with 15-row zero pad each side (+2 tail zero rows)
__global__ __launch_bounds__(256) void oaf_padrows(const bf16_t* __restrict__ src, bf16_t* __restrict__ dst, int KD){
  unsigned long long idx = (unsigned long long)blockIdx.x*256 + threadIdx.x; // 4*544*KD
  int k = (int)(idx % (unsigned)KD); unsigned long long rr = idx / (unsigned)KD;
  int p = (int)(rr % 544u); int b = (int)(rr / 544u);
  int ps = p - 15;
  float v = 0.f;
  if (ps >= 0 && ps < 512) v = (float)src[((unsigned long long)(b*512 + ps))*KD + k];
  dst[idx] = (bf16_t)v;
}

// comb = concat(onset, act) padded K 176->192, bf16 [2048,192]
__global__ __launch_bounds__(256) void oaf_comb(const float* __restrict__ onset, const float* __restrict__ act,
    bf16_t* __restrict__ dst){
  unsigned long long idx = (unsigned long long)blockIdx.x*256 + threadIdx.x; // 2048*192
  int k = (int)(idx % 192u); unsigned long long m = idx / 192u;
  float v = 0.f;
  if (k < 88) v = onset[m*88 + k];
  else if (k < 176) v = act[m*88 + (k-88)];
  dst[idx] = (bf16_t)v;
}

// windowed grouped attention: one wave per (b,t,g); lane = window tap (0..30)
__global__ __launch_bounds__(256) void oaf_attn(const float* __restrict__ q, const float* __restrict__ kbuf,
    const float* __restrict__ vbuf, const float* __restrict__ rel,
    bf16_t* __restrict__ outbf, float* __restrict__ aout){
  __shared__ float sE[8][32];
  int wave = threadIdx.x >> 5, lane = threadIdx.x & 31;
  int gid = blockIdx.x*8 + wave;            // (b*512+t)*8+g
  int g = gid & 7; int bt = gid >> 3; int t = bt & 511; int b = bt >> 9;
  const float* qp = q + (unsigned long long)bt*768 + g*96;
  unsigned long long base = ((unsigned long long)(b*544 + t))*768 + (unsigned)(g*96);
  float e = -3.0e38f;
  if (lane < 31){
    const float* kp = kbuf + base + (unsigned long long)lane*768;
    const float* rp = rel + (g*96)*31 + lane;   // rel[(g*96+d)*31 + lane]
    float s = 0.f;
    for (int d = 0; d < 96; d++) s += qp[d] * (kp[d] + rp[d*31]);
    e = s;
  }
  sE[wave][lane] = e;
  __syncthreads();
  float mx = -3.0e38f;
  for (int kk=0; kk<31; kk++) mx = fmaxf(mx, sE[wave][kk]);
  float ssum = 0.f;
  for (int kk=0; kk<31; kk++) ssum += __expf(sE[wave][kk] - mx);
  float aw = (lane < 31) ? __expf(e - mx) / ssum : 0.f;
  __syncthreads();
  sE[wave][lane] = aw;
  __syncthreads();
  if (aout != nullptr && lane < 31) aout[(unsigned long long)gid*31 + lane] = aw;
  float o0=0.f, o1=0.f, o2=0.f;
  for (int kk=0; kk<31; kk++){
    float w = sE[wave][kk];
    const float* vp = vbuf + base + (unsigned long long)kk*768;
    o0 += w*vp[lane]; o1 += w*vp[lane+32]; o2 += w*vp[lane+64];
  }
  bf16_t* op = outbf + (unsigned long long)bt*768 + g*96;
  op[lane]=(bf16_t)o0; op[lane+32]=(bf16_t)o1; op[lane+64]=(bf16_t)o2;
}

// ---------------- host ----------------

struct CS { const float *w1,*b1,*w2,*b2,*w3,*b3,*fcw,*fcb,*g1,*be1,*m1,*v1,*g2,*be2,*m2,*v2,*g3,*be3,*m3,*v3; };
struct AT { const float *Wq,*Wk,*Wv,*rel; };
struct LN { const float *w,*b; };

extern "C" void kernel_launch(void* const* d_in, const int* in_sizes, int n_in,
                              void* d_out, int out_size, void* d_ws, size_t ws_size,
                              hipStream_t stream) {
  (void)out_size; (void)ws_size;
  const float* P[64];
  for (int i = 0; i < n_in && i < 64; i++) P[i] = (const float*)d_in[i];

  CS oc, fcv; AT oa, ca; LN ol, fl, cl; const float* spec;
  if (in_sizes[0] == 468992) {
    // insertion order (depth-first)
    spec = P[0];
    auto cs = [&](int b)->CS{ return {P[b],P[b+1],P[b+2],P[b+3],P[b+4],P[b+5],P[b+6],P[b+7],
                                      P[b+8],P[b+9],P[b+10],P[b+11],P[b+12],P[b+13],P[b+14],P[b+15],
                                      P[b+16],P[b+17],P[b+18],P[b+19]}; };
    oc = cs(1);  oa = {P[21],P[22],P[23],P[24]}; ol = {P[25],P[26]};
    fcv = cs(27); fl = {P[47],P[48]}; ca = {P[49],P[50],P[51],P[52]}; cl = {P[53],P[54]};
  } else {
    // jax tree_flatten alphabetical order
    auto cs = [&](int b)->CS{ return {P[b+17],P[b+0],P[b+18],P[b+1],P[b+19],P[b+2],P[b+7],P[b+6],
                                      P[b+8],P[b+3],P[b+11],P[b+14],P[b+9],P[b+4],P[b+12],P[b+15],
                                      P[b+10],P[b+5],P[b+13],P[b+16]}; };
    ca = {P[1],P[0],P[2],P[3]}; cl = {P[5],P[4]};
    fcv = cs(6); fl = {P[27],P[26]};
    oa = {P[29],P[28],P[30],P[31]};
    oc = cs(32); ol = {P[53],P[52]};
    spec = P[54];
  }

  float* out = (float*)d_out;
  float* out_onset = out;                 // 2048*88
  float* out_act   = out + 180224;        // 2048*88
  float* out_frame = out + 360448;        // 2048*88
  float* out_ac    = out + 540672;        // 2048*8*31

  // workspace bump allocator
  size_t off = 0;
  auto alloc = [&](size_t bytes)->char*{
    off = (off + 255) & ~(size_t)255;
    char* p = (char*)d_ws + off; off += bytes; return p;
  };
  float* bnA1 = (float*)alloc(96*4); float* bnB1 = (float*)alloc(96*4);
  float* bnA2 = (float*)alloc(96*4); float* bnB2 = (float*)alloc(96*4);
  float* bnA3 = (float*)alloc(96*4); float* bnB3 = (float*)alloc(96*4);
  bf16_t* convw2s = (bf16_t*)alloc(9ULL*64*64*2);
  bf16_t* convw3s = (bf16_t*)alloc(9ULL*96*64*2);
  bf16_t* fcws    = (bf16_t*)alloc(768ULL*5472*2);
  bf16_t* act1    = (bf16_t*)alloc(468992ULL*64*2);   // also reused as act3 [B,512,114,96]
  bf16_t* act2    = (bf16_t*)alloc(468992ULL*64*2);   // also reused as A_fc [2048,5472]
  bf16_t* pool1b  = (bf16_t*)alloc(4ULL*512*114*64*2);
  bf16_t* h_on    = (bf16_t*)alloc(2048ULL*768*2);
  bf16_t* h_fr    = (bf16_t*)alloc(2048ULL*768*2);
  bf16_t* Wqs     = (bf16_t*)alloc(768ULL*768*2);
  bf16_t* Wks     = (bf16_t*)alloc(768ULL*768*2);
  bf16_t* Wvs     = (bf16_t*)alloc(768ULL*768*2);
  bf16_t* lins    = (bf16_t*)alloc(128ULL*768*2);
  bf16_t* xp      = (bf16_t*)alloc(2176ULL*768*2);    // reused for comb xp [2176,192]
  float*  qf      = (float*)alloc(2048ULL*768*4);
  float*  kf      = (float*)alloc(2176ULL*768*4);
  float*  vf      = (float*)alloc(2176ULL*768*4);
  bf16_t* attnout = (bf16_t*)alloc(2048ULL*768*2);
  bf16_t* combbf  = (bf16_t*)alloc(2048ULL*192*2);
  bf16_t* act3    = act1;
  bf16_t* A_fc    = act2;

  auto gemm = [&](const bf16_t* A, const bf16_t* W, const float* bias,
                  float* oF, int ldoF, bf16_t* oB, int ldoB,
                  int M, int N, int Npad, int K, int act){
    dim3 g(M/128, Npad/64);
    oaf_gemm<<<g, 256, 0, stream>>>(A, W, bias, oF, ldoF, oB, ldoB, N, K, act);
  };
  auto stagew = [&](const float* src, bf16_t* dst, int N, int Ksrc, int Npad, int Kp){
    unsigned long long total = (unsigned long long)Npad * Kp;
    oaf_stagew<<<(unsigned)((total + 255)/256), 256, 0, stream>>>(src, dst, N, Ksrc, Kp, total);
  };

  auto run_stack = [&](const CS& c, bf16_t* h_bf){
    oaf_bnprep<<<1,128,0,stream>>>(c.g1,c.be1,c.m1,c.v1,c.b1,bnA1,bnB1,48);
    oaf_bnprep<<<1,128,0,stream>>>(c.g2,c.be2,c.m2,c.v2,c.b2,bnA2,bnB2,48);
    oaf_bnprep<<<1,128,0,stream>>>(c.g3,c.be3,c.m3,c.v3,c.b3,bnA3,bnB3,96);
    oaf_stagecw<<<144,256,0,stream>>>(c.w2, convw2s, 48, 48, 64);
    oaf_stagecw<<<216,256,0,stream>>>(c.w3, convw3s, 96, 48, 96);
    stagew(c.fcw, fcws, 768, 5472, 768, 5472);
    oaf_conv1<<<117248,256,0,stream>>>(spec, c.w1, bnA1, bnB1, act1);
    oaf_conv3x3<<<dim3(7328,2),256,0,stream>>>(act1, convw2s, bnA2, bnB2, act2, 512, 229, 48, 64);
    oaf_pool<<<58368,256,0,stream>>>(act2, pool1b, 229, 114, 64);
    oaf_conv3x3<<<dim3(3648,3),256,0,stream>>>(pool1b, convw3s, bnA3, bnB3, act3, 512, 114, 96, 96);
    oaf_pool2<<<43776,256,0,stream>>>(act3, A_fc);
    gemm(A_fc, fcws, c.fcb, nullptr, 0, h_bf, 768, 2048, 768, 768, 5472, 0);
  };

  // ---- onset & frame conv stacks (sequential, shared scratch) ----
  run_stack(oc, h_on);
  run_stack(fcv, h_fr);

  // ---- onset local attention ----
  stagew(oa.Wq, Wqs, 768, 768, 768, 768);
  stagew(oa.Wk, Wks, 768, 768, 768, 768);
  stagew(oa.Wv, Wvs, 768, 768, 768, 768);
  oaf_padrows<<<6528,256,0,stream>>>(h_on, xp, 768);
  gemm(h_on, Wqs, nullptr, qf, 768, nullptr, 0, 2048, 768, 768, 768, 0);
  gemm(xp,   Wks, nullptr, kf, 768, nullptr, 0, 2176, 768, 768, 768, 0);
  gemm(xp,   Wvs, nullptr, vf, 768, nullptr, 0, 2176, 768, 768, 768, 0);
  oaf_attn<<<2048,256,0,stream>>>(qf, kf, vf, oa.rel, attnout, nullptr);

  // onset = sigmoid(lin(attn_out))
  stagew(ol.w, lins, 88, 768, 128, 768);
  gemm(attnout, lins, ol.b, out_onset, 88, nullptr, 0, 2048, 88, 128, 768, 1);

  // act = sigmoid(lin(h_frame))
  stagew(fl.w, lins, 88, 768, 128, 768);
  gemm(h_fr, lins, fl.b, out_act, 88, nullptr, 0, 2048, 88, 128, 768, 1);

  // ---- comb attention ----
  oaf_comb<<<1536,256,0,stream>>>(out_onset, out_act, combbf);
  oaf_padrows<<<1632,256,0,stream>>>(combbf, xp, 192);
  stagew(ca.Wq, Wqs, 768, 176, 768, 192);
  stagew(ca.Wk, Wks, 768, 176, 768, 192);
  stagew(ca.Wv, Wvs, 768, 176, 768, 192);
  gemm(combbf, Wqs, nullptr, qf, 768, nullptr, 0, 2048, 768, 768, 192, 0);
  gemm(xp,     Wks, nullptr, kf, 768, nullptr, 0, 2176, 768, 768, 192, 0);
  gemm(xp,     Wvs, nullptr, vf, 768, nullptr, 0, 2176, 768, 768, 192, 0);
  oaf_attn<<<2048,256,0,stream>>>(qf, kf, vf, ca.rel, attnout, out_ac);

  // frame = sigmoid(lin(comb_attn_out))
  stagew(cl.w, lins, 88, 768, 128, 768);
  gemm(attnout, lins, cl.b, out_frame, 88, nullptr, 0, 2048, 88, 128, 768, 1);
}
